// _ResidualQ_29609504538633
// MI455X (gfx1250) — matrix operand order repaired and checked by reading, not compiled, not run on hardware
//
#include <hip/hip_runtime.h>

// ---------------- problem constants ----------------
#define BATCH 8
#define TT    4096
#define DIN   1024
#define DR    512
#define DOUT  1024
#define NQn   8
#define KCB   2048
#define DC    256

// ---------------- GEMM tiling ----------------
#define BM 128          // block tile M (4 waves of 32 rows)
#define BN 64           // block tile N (2 waves of 32 cols)
#define BK 32           // one wmma K per step

typedef __attribute__((ext_vector_type(16))) __bf16 v16bf;
typedef __attribute__((ext_vector_type(8)))  float  v8f;

union ABFrag { v16bf v; unsigned int u[8]; };

__device__ __forceinline__ unsigned short f2bf(float f) {
  union { float f; unsigned int u; } x; x.f = f;
  unsigned int r = x.u + 0x7FFFu + ((x.u >> 16) & 1u);   // round-nearest-even
  return (unsigned short)(r >> 16);
}

// A fragment (16x32 bf16, M x K), CDNA5 16-bit A layout:
// lane l: row = l%16; h=l/16; VGPR j<4 -> K = 8h+2j,+1 ; j>=4 -> K = 16+8h+2(j-4),+1
// (contiguous in groups of 4 VGPRs -> merges to b128 loads)
__device__ __forceinline__ v16bf load_a_frag(const unsigned short* p0, int lane, int ldk, int k0) {
  ABFrag a; const int h = lane >> 4;
  const unsigned short* p = p0 + (size_t)(lane & 15) * ldk + k0;
#pragma unroll
  for (int j = 0; j < 8; ++j) {
    int k = (j < 4) ? (8 * h + 2 * j) : (16 + 8 * h + 2 * (j - 4));
    a.u[j] = *(const unsigned int*)(p + k);
  }
  return a.v;
}

// B fragment (32x16 bf16, K x N) from LDS stored [n][k]: lane l: n=l%16, same K order as A fragment
__device__ __forceinline__ v16bf load_b_frag(const unsigned short* p0, int lane, int ldk, int k0) {
  ABFrag b; const int h = lane >> 4;
  const unsigned short* p = p0 + (size_t)(lane & 15) * ldk + k0;
#pragma unroll
  for (int j = 0; j < 8; ++j) b.u[j] = *(const unsigned int*)(p + ((j < 4) ? (8 * h + 2 * j) : (16 + 8 * h + 2 * (j - 4))));
  return b.v;
}

__device__ __forceinline__ v8f wmma_bf16(v16bf a, v16bf b, v8f c) {
  return __builtin_amdgcn_wmma_f32_16x16x32_bf16(false, a, false, b, (short)0, c, false, false);
}

// ---------------- helper kernels ----------------
__global__ void k_cvt_bf16(const float* __restrict__ in, unsigned short* __restrict__ out, int n) {
  int i = blockIdx.x * blockDim.x + threadIdx.x;
  if (i < n) out[i] = f2bf(in[i]);
}

// one block (256 thr) per codebook row: normalized bf16 codebook
__global__ void k_cbnorm(const float* __restrict__ cb, unsigned short* __restrict__ cbn) {
  __shared__ float red[256];
  const int row = blockIdx.x;
  const float v = cb[(size_t)row * DC + threadIdx.x];
  red[threadIdx.x] = v * v;
  __syncthreads();
  for (int s = 128; s > 0; s >>= 1) {
    if (threadIdx.x < s) red[threadIdx.x] += red[threadIdx.x + s];
    __syncthreads();
  }
  const float sc = 1.0f / fmaxf(sqrtf(red[0]), 1e-12f);
  cbn[(size_t)row * DC + threadIdx.x] = f2bf(v * sc);
}

__global__ void k_tail(const int* __restrict__ lengths, float* __restrict__ dst) {
  if (threadIdx.x < BATCH) dst[threadIdx.x] = (float)lengths[threadIdx.x];
}

// ---------------- unified double-buffered WMMA GEMM ----------------
// C(Md x TT) = Wb(Md x Kd) @ B(Kd x TT) + bias, per batch b = blockIdx.z
// BMODE: 0 = B from f32 activation (D,T)  1 = same + zero columns t>=len[b]
//        2 = B gathered from codebook rows cb[ids[b,t]]
// OMODE: 0 = residual <- C ; out_acc <- 0            1 = zE(bf16) <- C
//        2 = masked: out_acc += C ; residual -= C    3 = quant <- C
template <int Md, int Kd, int BMODE, int OMODE>
__global__ void __launch_bounds__(256) k_gemm(
    const unsigned short* __restrict__ Wb, const float* __restrict__ bias,
    const float* __restrict__ Bact, const int* __restrict__ lengths,
    const int* __restrict__ ids, const float* __restrict__ cb,
    float* __restrict__ o_res, float* __restrict__ o_acc,
    unsigned short* __restrict__ o_bf, float* __restrict__ o_f32) {
  constexpr int NKT = Kd / BK;
  __shared__ alignas(16) unsigned short sA[2][BM * BK];   // 2 x 8KB
  __shared__ alignas(16) unsigned short sB[2][BN * BK];   // 2 x 4KB
  const int b = blockIdx.z, m0 = blockIdx.y * BM, t0 = blockIdx.x * BN;
  const int tid = threadIdx.x, lane = tid & 31, wid = tid >> 5;
  const int wm = wid >> 1, wn = wid & 1;                  // 4 x 2 waves, 32x32 each

  int L = TT;
  if constexpr (BMODE == 1 || OMODE == 2) L = lengths[b];

  // A staging: 128x32 bf16 tile, each thread copies 16 contiguous bf16 (32B)
  const int a_idx = tid * 16;
  const int a_r = a_idx >> 5, a_c = a_idx & 31;
  const unsigned short* a_src = Wb + (size_t)(m0 + a_r) * Kd + a_c;

  // B staging indices
  const float* b_src = nullptr;
  int b_kr = 0, b_cc = 0, g_tcol = 0, g_c0 = 0, myId = 0;
  if constexpr (BMODE != 2) {
    const int idx = tid * 8;
    b_kr = idx >> 6; b_cc = idx & 63;                     // 8 consecutive t of one k-row
    b_src = Bact + (size_t)b * Kd * TT + (size_t)b_kr * TT + t0 + b_cc;
  } else {
    g_tcol = tid & 63; g_c0 = (tid >> 6) * 8;             // 8 consecutive c of one column
    myId = ids[(size_t)b * TT + t0 + g_tcol];             // gather index, hoisted
  }

  uint4 ar0, ar1; float4 br0, br1;                        // prefetch registers

  auto gload = [&](int k0) {
    ar0 = *(const uint4*)(a_src + k0);
    ar1 = *(const uint4*)(a_src + k0 + 8);
    if constexpr (BMODE != 2) {
      const float* s = b_src + (size_t)k0 * TT;
      br0 = *(const float4*)s; br1 = *(const float4*)(s + 4);
    } else {
      const float* s = cb + (size_t)myId * Kd + k0 + g_c0;
      br0 = *(const float4*)s; br1 = *(const float4*)(s + 4);
    }
  };
  auto lstore = [&](int buf) {
    *(uint4*)&sA[buf][a_idx] = ar0;
    *(uint4*)&sA[buf][a_idx + 8] = ar1;
    float f[8] = {br0.x, br0.y, br0.z, br0.w, br1.x, br1.y, br1.z, br1.w};
    if constexpr (BMODE != 2) {
#pragma unroll
      for (int i = 0; i < 8; ++i) {
        float fv = f[i];
        if constexpr (BMODE == 1) fv = ((t0 + b_cc + i) < L) ? fv : 0.0f;
        sB[buf][(b_cc + i) * BK + b_kr] = f2bf(fv);       // transpose to [n][k]
      }
    } else {
      union { uint4 q; unsigned short h[8]; } u;
#pragma unroll
      for (int i = 0; i < 8; ++i) u.h[i] = f2bf(f[i]);
      *(uint4*)&sB[buf][g_tcol * BK + g_c0] = u.q;        // already [n][k]
    }
  };

  v8f acc00 = {}, acc01 = {}, acc10 = {}, acc11 = {};
  gload(0);
  lstore(0);
  __syncthreads();
  int cur = 0;
  for (int ks = 0; ks < NKT; ++ks) {
    if (ks + 1 < NKT) gload((ks + 1) * BK);               // prefetch overlaps wmma
    v16bf a0 = load_a_frag(&sA[cur][(wm * 32 + 0)  * BK], lane, BK, 0);
    v16bf a1 = load_a_frag(&sA[cur][(wm * 32 + 16) * BK], lane, BK, 0);
    v16bf b0 = load_b_frag(&sB[cur][(wn * 32 + 0)  * BK], lane, BK, 0);
    v16bf b1 = load_b_frag(&sB[cur][(wn * 32 + 16) * BK], lane, BK, 0);
    acc00 = wmma_bf16(a0, b0, acc00);
    acc01 = wmma_bf16(a0, b1, acc01);
    acc10 = wmma_bf16(a1, b0, acc10);
    acc11 = wmma_bf16(a1, b1, acc11);
    if (ks + 1 < NKT) lstore(cur ^ 1);                    // fill other buffer
    __syncthreads();                                      // single barrier per K-step
    cur ^= 1;
  }

  const int h = lane >> 4;
  const int nb = t0 + wn * 32 + (lane & 15);
  const int mbase = m0 + wm * 32 + 8 * h;
  v8f accs[2][2] = {{acc00, acc01}, {acc10, acc11}};
#pragma unroll
  for (int mi = 0; mi < 2; ++mi) {
#pragma unroll
    for (int ni = 0; ni < 2; ++ni) {
      const int n = nb + ni * 16;
      v8f a = accs[mi][ni];
#pragma unroll
      for (int r = 0; r < 8; ++r) {
        const int m = mbase + mi * 16 + r;
        const float v = a[r] + bias[m];
        const size_t o = (size_t)b * Md * TT + (size_t)m * TT + n;
        if constexpr (OMODE == 0)      { o_res[o] = v; o_acc[o] = 0.0f; }
        else if constexpr (OMODE == 1) { o_bf[o] = f2bf(v); }
        else if constexpr (OMODE == 2) { if (n < L) { o_acc[o] += v; o_res[o] -= v; } }
        else                           { o_f32[o] = v; }
      }
    }
  }
}

// ---------------- code search: ids = argmax_k <z_e, cb_n[k]> (== argmin dist) ----------------
// B-fragments (one 16-t strip of z_e, full Dc) live in VGPRs across all 128 codebook
// tiles; A-fragments are double-buffered ACROSS tiles so the 16 global_load_b128 of
// tile mt+8 are in flight while the 8-WMMA chain of tile mt executes.
__global__ void __launch_bounds__(256, 1) k_score(
    const unsigned short* __restrict__ cbn, const unsigned short* __restrict__ zE,
    int* __restrict__ ids, float* __restrict__ codes) {
  __shared__ alignas(16) unsigned short sB[16 * DC];   // [n][k], 8 KB
  __shared__ float sVal[8 * 16];
  __shared__ int   sIdx[8 * 16];
  const int b = blockIdx.z, t0 = blockIdx.x * 16;
  const int tid = threadIdx.x, lane = tid & 31, wid = tid >> 5;
  // stage one 16-column strip of z_e (full Dc) into LDS, transposed to [n][k]
  {
    const unsigned short* s = zE + (size_t)b * DC * TT + (size_t)tid * TT + t0;
    union { uint4 q; unsigned short h[8]; } u0, u1;
    u0.q = *(const uint4*)s;
    u1.q = *(const uint4*)(s + 8);
#pragma unroll
    for (int n = 0; n < 8; ++n) sB[n * DC + tid] = u0.h[n];
#pragma unroll
    for (int n = 0; n < 8; ++n) sB[(n + 8) * DC + tid] = u1.h[n];
  }
  __syncthreads();
  v16bf bf[8];
#pragma unroll
  for (int s = 0; s < 8; ++s) bf[s] = load_b_frag(sB, lane, DC, s * 32);

  const int h = lane >> 4;
  float bestV = -3.402823466e38f;
  int   bestI = 0x7fffffff;

  v16bf afA[8], afB[8];
  auto loadfrags = [&](v16bf* af, int mt) {
    const unsigned short* arow = cbn + (size_t)mt * 16 * DC;
#pragma unroll
    for (int s = 0; s < 8; ++s) af[s] = load_a_frag(arow, lane, DC, s * 32);
  };
  auto process = [&](const v16bf* af, int mt) {
    v8f acc = {};
#pragma unroll
    for (int s = 0; s < 8; ++s) acc = wmma_bf16(af[s], bf[s], acc);
#pragma unroll
    for (int r = 0; r < 8; ++r) {               // branchless argmax, first-index ties
      float v = acc[r];
      int m = mt * 16 + 8 * h + r;
      bool upd = (v > bestV) || ((v == bestV) && (m < bestI));
      bestV = upd ? v : bestV;
      bestI = upd ? m : bestI;
    }
  };

  constexpr int NT = KCB / 16 / 8;              // 16 tiles per wave
  loadfrags(afA, wid);
#pragma unroll
  for (int i = 0; i < NT; i += 2) {
    const int mt0 = wid + i * 8, mt1 = mt0 + 8;
    if (i + 1 < NT) loadfrags(afB, mt1);        // in flight during afA's wmma chain
    process(afA, mt0);
    if (i + 2 < NT) loadfrags(afA, mt0 + 16);   // in flight during afB's wmma chain
    if (i + 1 < NT) process(afB, mt1);
  }

  {  // combine lane l with lane l^16 (same column, other row half)
    float oV = __shfl_xor(bestV, 16, 32);
    int   oI = __shfl_xor(bestI, 16, 32);
    bool upd = (oV > bestV) || ((oV == bestV) && (oI < bestI));
    bestV = upd ? oV : bestV;
    bestI = upd ? oI : bestI;
  }
  if (lane < 16) { sVal[wid * 16 + lane] = bestV; sIdx[wid * 16 + lane] = bestI; }
  __syncthreads();
  if (tid < 16) {
    float v = sVal[tid]; int i = sIdx[tid];
#pragma unroll
    for (int w = 1; w < 8; ++w) {
      float v2 = sVal[w * 16 + tid]; int i2 = sIdx[w * 16 + tid];
      bool upd = (v2 > v) || ((v2 == v) && (i2 < i));
      v = upd ? v2 : v;
      i = upd ? i2 : i;
    }
    const int t = t0 + tid;
    ids[(size_t)b * TT + t] = i;
    codes[(size_t)b * TT + t] = (float)i;
  }
}

// ---------------- host ----------------
extern "C" void kernel_launch(void* const* d_in, const int* in_sizes, int n_in,
                              void* d_out, int out_size, void* d_ws, size_t ws_size,
                              hipStream_t stream) {
  const float* z         = (const float*)d_in[0];
  const int*   lengths   = (const int*)d_in[1];
  const float* ip_w      = (const float*)d_in[2];
  const float* ip_b      = (const float*)d_in[3];
  const float* op_w      = (const float*)d_in[4];
  const float* op_b      = (const float*)d_in[5];
  const float* q_in_w    = (const float*)d_in[6];
  const float* q_in_b    = (const float*)d_in[7];
  const float* q_out_w   = (const float*)d_in[8];
  const float* q_out_b   = (const float*)d_in[9];
  const float* codebooks = (const float*)d_in[10];

  char* p = (char*)d_ws;
  auto take = [&](size_t bytes) { char* r = p; p += (bytes + 255) & ~(size_t)255; return r; };
  float*          residual = (float*)take((size_t)BATCH * DR * TT * 4);
  float*          out_acc  = (float*)take((size_t)BATCH * DR * TT * 4);
  unsigned short* zE       = (unsigned short*)take((size_t)BATCH * DC * TT * 2);
  int*            ids      = (int*)take((size_t)BATCH * TT * 4);
  unsigned short* ipw      = (unsigned short*)take((size_t)DR * DIN * 2);
  unsigned short* opw      = (unsigned short*)take((size_t)DOUT * DR * 2);
  unsigned short* qinw     = (unsigned short*)take((size_t)NQn * DC * DR * 2);
  unsigned short* qoutw    = (unsigned short*)take((size_t)NQn * DR * DC * 2);
  unsigned short* cbn      = (unsigned short*)take((size_t)NQn * KCB * DC * 2);

  // bf16 weight conversions + normalized codebooks
  k_cvt_bf16<<<(DR * DIN) / 256, 256, 0, stream>>>(ip_w, ipw, DR * DIN);
  k_cvt_bf16<<<(DOUT * DR) / 256, 256, 0, stream>>>(op_w, opw, DOUT * DR);
  k_cvt_bf16<<<(NQn * DC * DR) / 256, 256, 0, stream>>>(q_in_w, qinw, NQn * DC * DR);
  k_cvt_bf16<<<(NQn * DR * DC) / 256, 256, 0, stream>>>(q_out_w, qoutw, NQn * DR * DC);
  k_cbnorm<<<NQn * KCB, 256, 0, stream>>>(codebooks, cbn);

  float* quant = (float*)d_out;
  float* codes = quant + (size_t)BATCH * DOUT * TT;

  // residual0 = ip_w @ z + ip_b ; out_acc = 0
  k_gemm<DR, DIN, 0, 0><<<dim3(TT / BN, DR / BM, BATCH), 256, 0, stream>>>(
      ipw, ip_b, z, nullptr, nullptr, nullptr, residual, out_acc, nullptr, nullptr);

  for (int q = 0; q < NQn; ++q) {
    // z_e = q_in_w @ (residual * mask) + b   (bf16)
    k_gemm<DC, DR, 1, 1><<<dim3(TT / BN, DC / BM, BATCH), 256, 0, stream>>>(
        qinw + (size_t)q * DC * DR, q_in_b + q * DC, residual, lengths,
        nullptr, nullptr, nullptr, nullptr, zE, nullptr);
    // nearest-code search
    k_score<<<dim3(TT / 16, 1, BATCH), 256, 0, stream>>>(
        cbn + (size_t)q * KCB * DC, zE, ids, codes + (size_t)q * BATCH * TT);
    // zq = q_out_w @ cb[ids] + b ; out += zq*mask ; residual -= zq*mask
    k_gemm<DR, DC, 2, 2><<<dim3(TT / BN, DR / BM, BATCH), 256, 0, stream>>>(
        qoutw + (size_t)q * DR * DC, q_out_b + q * DR, nullptr, lengths,
        ids, codebooks + (size_t)q * KCB * DC, residual, out_acc, nullptr, nullptr);
  }

  // quantized = op_w @ out + op_b
  k_gemm<DOUT, DR, 0, 3><<<dim3(TT / BN, DOUT / BM, BATCH), 256, 0, stream>>>(
      opw, op_b, out_acc, nullptr, nullptr, nullptr, nullptr, nullptr, nullptr, quant);

  k_tail<<<1, 32, 0, stream>>>(lengths, codes + (size_t)NQn * BATCH * TT);
}
